// ImageGNN_48369921687741
// MI455X (gfx1250) — matrix operand order repaired and checked by reading, not compiled, not run on hardware
//
#include <hip/hip_runtime.h>
#include <hip/hip_bf16.h>

// ---------------- problem constants ----------------
#define Bn    8
#define Cc    96
#define Nn    3136          // 56*56
#define Kk    9
#define COUTc 192
#define MNODES (Bn * Nn)    // 25088, divisible by 16

typedef __attribute__((ext_vector_type(16))) __bf16 v16bf;
typedef __attribute__((ext_vector_type(8)))  float  v8f;

#define WMMA_BF16(A_, B_, C_) \
  __builtin_amdgcn_wmma_f32_16x16x32_bf16(false, (A_), false, (B_), (short)0, (C_), false, false)

// ---------------- workspace layout ----------------
__host__ __device__ constexpr size_t alignup(size_t x) { return (x + 255) & ~size_t(255); }

constexpr size_t O_XF    = 0;                                      // bf16 [MNODES,96]
constexpr size_t O_SQ    = alignup(O_XF    + (size_t)MNODES*96*2); // f32  [MNODES]
constexpr size_t O_NN    = alignup(O_SQ    + (size_t)MNODES*4);    // i32  [MNODES,9]
constexpr size_t O_KI    = alignup(O_NN    + (size_t)MNODES*9*4);  // i32  [MNODES]
constexpr size_t O_KF    = alignup(O_KI    + (size_t)MNODES*4);    // f32  [MNODES]
// all weights stored TRANSPOSED as [N][K] bf16 (K padded to mult of 32)
constexpr size_t O_WKMAP = alignup(O_KF    + (size_t)MNODES*4);    // [512][96]
constexpr size_t O_WKFC  = alignup(O_WKMAP + (size_t)512*96*2);    // [64][512]
constexpr size_t O_WKMU  = alignup(O_WKFC  + (size_t)64*512*2);    // [32][64]
constexpr size_t O_WEC1  = alignup(O_WKMU  + (size_t)32*64*2);     // [96][192]
constexpr size_t O_WEC2  = alignup(O_WEC1  + (size_t)96*192*2);    // [96][192]
constexpr size_t O_WFC   = alignup(O_WEC2  + (size_t)96*192*2);    // [192][96]
constexpr size_t O_WIO   = alignup(O_WFC   + (size_t)192*96*2);    // [192][96]
constexpr size_t O_WUP   = alignup(O_WIO   + (size_t)192*96*2);    // [192][384]
constexpr size_t O_HK    = alignup(O_WUP   + (size_t)192*384*2);   // bf16 [MNODES,512]
constexpr size_t O_H     = alignup(O_HK    + (size_t)MNODES*512*2);// bf16 [MNODES,64]
constexpr size_t O_MU    = alignup(O_H     + (size_t)MNODES*64*2); // bf16 [MNODES,32]
constexpr size_t O_E1    = alignup(O_MU    + (size_t)MNODES*32*2); // bf16 [MNODES,192]
constexpr size_t O_H1    = alignup(O_E1    + (size_t)MNODES*192*2);// bf16 [MNODES,96]
constexpr size_t O_E2    = alignup(O_H1    + (size_t)MNODES*96*2); // bf16 [MNODES,192]
constexpr size_t O_H2    = alignup(O_E2    + (size_t)MNODES*192*2);// bf16 [MNODES,96]
constexpr size_t O_XA    = alignup(O_H2    + (size_t)MNODES*96*2); // bf16 [MNODES,384]

// ---------------- fragment loaders ----------------
// A fragment (16x32, 16-bit), row-major [M,K]: lane half h in {0,1}:
//   elements 0..15 -> K = k0 + 16h + t  (contiguous, matches Bt)
__device__ inline v16bf load_frag_A(const __bf16* __restrict__ A, int lda, int row,
                                    int k0, int half) {
  v16bf a;
  const __bf16* p = A + (size_t)row * lda + k0 + 16 * half;
#pragma unroll
  for (int t = 0; t < 16; ++t) a[t] = p[t];
  return a;
}

// B fragment (32x16, 16-bit) from a matrix stored [N][K] row-major (i.e. B transposed):
//   lane = col(0..15) + 16*half; elements 0..15 -> K = k0 + 16*half + t  (contiguous!)
__device__ inline v16bf load_frag_Bt(const __bf16* __restrict__ Bt, int ldb, int col,
                                     int k0, int half) {
  v16bf b;
  const __bf16* p = Bt + (size_t)col * ldb + k0 + 16 * half;
#pragma unroll
  for (int t = 0; t < 16; ++t) b[t] = p[t];
  return b;
}

// ---------------- prep kernels ----------------
__global__ void pack_xf_kernel(const float* __restrict__ x, __bf16* __restrict__ xfb) {
  size_t tid = (size_t)blockIdx.x * 256 + threadIdx.x;       // over B*C*N (x layout)
  if (tid >= (size_t)Bn * Cc * Nn) return;
  int n = (int)(tid % Nn);
  size_t t = tid / Nn;
  int c = (int)(t % Cc);
  int b = (int)(t / Cc);
  xfb[((size_t)(b * Nn + n)) * Cc + c] = (__bf16)x[tid];
}

__global__ void sqnorm_kernel(const __bf16* __restrict__ xfb, float* __restrict__ sq) {
  int node = blockIdx.x;
  int lane = threadIdx.x;
  float s = 0.f;
  for (int c = lane; c < Cc; c += 32) {
    float v = (float)xfb[(size_t)node * Cc + c];
    s += v * v;
  }
#pragma unroll
  for (int off = 16; off; off >>= 1) s += __shfl_down(s, off, 32);
  if (lane == 0) sq[node] = s;
}

// convert f32 [R,C] -> bf16 TRANSPOSED+padded [Cp][Rp] (zero pad)
__global__ void padT_bf16_kernel(const float* __restrict__ src, __bf16* __restrict__ dst,
                                 int R, int C_, int Rp, int Cp) {
  int tid = blockIdx.x * 256 + threadIdx.x;
  if (tid >= Rp * Cp) return;
  int c = tid / Rp, r = tid % Rp;   // dst[c][r], r is contiguous (coalesced writes)
  float v = (r < R && c < C_) ? src[r * C_ + c] : 0.f;
  dst[tid] = (__bf16)v;
}

// ---------------- kNN top-9 via WMMA Gram tiles ----------------
__global__ void knn_topk_kernel(const __bf16* __restrict__ xfb, const float* __restrict__ sq,
                                int* __restrict__ nn) {
  __shared__ float tile[16][17];
  __shared__ float sqrow[16];
  int lane = threadIdx.x & 31;
  int half = lane >> 4, lidx = lane & 15;
  int base = blockIdx.y * Nn;
  int rowt = blockIdx.x * 16;
  int grow = base + rowt + lidx;

  if (lane < 16) sqrow[lane] = sq[base + rowt + lane];
  __syncthreads();

  v16bf a0 = load_frag_A(xfb, Cc, grow, 0, half);
  v16bf a1 = load_frag_A(xfb, Cc, grow, 32, half);
  v16bf a2 = load_frag_A(xfb, Cc, grow, 64, half);

  float best[Kk];
  int   bidx[Kk];
#pragma unroll
  for (int e = 0; e < Kk; ++e) { best[e] = 3.0e38f; bidx[e] = base; }

  for (int ct = 0; ct < Nn / 16; ++ct) {
    int colg = base + ct * 16 + lidx;
    // xfb is [node][K] -> exactly the Bt layout, contiguous per lane
    v16bf b0 = load_frag_Bt(xfb, Cc, colg, 0, half);
    v16bf b1 = load_frag_Bt(xfb, Cc, colg, 32, half);
    v16bf b2 = load_frag_Bt(xfb, Cc, colg, 64, half);
    if (ct + 1 < Nn / 16)
      __builtin_prefetch((const void*)(xfb + (size_t)(colg + 16) * Cc), 0, 1);
    v8f acc = {};
    acc = WMMA_BF16(a0, b0, acc);
    acc = WMMA_BF16(a1, b1, acc);
    acc = WMMA_BF16(a2, b2, acc);
    float sqc = sq[colg];

    __syncthreads();  // previous tile fully consumed
#pragma unroll
    for (int r = 0; r < 8; ++r) {
      int m = r + 8 * half;
      tile[m][lidx] = sqrow[m] - 2.0f * acc[r] + sqc;
    }
    __syncthreads();

    if (lane < 16) {
      for (int cc = 0; cc < 16; ++cc) {
        float d = tile[lane][cc];
        if (d < best[Kk - 1]) {
          int gidx = base + ct * 16 + cc;
          int pos = Kk - 1;
#pragma unroll
          for (int e = Kk - 2; e >= 0; --e) {
            if (d < best[e]) { best[e + 1] = best[e]; bidx[e + 1] = bidx[e]; pos = e; }
          }
          best[pos] = d; bidx[pos] = gidx;
        }
      }
    }
  }
  if (lane < 16) {
    int row = base + rowt + lane;
#pragma unroll
    for (int e = 0; e < Kk; ++e) nn[row * Kk + e] = bidx[e];
  }
}

// ---------------- generic WMMA GEMM: C = A@Bt^T (+ rowscale*bias) [relu] ----------------
// A: bf16 [M,K] row-major.  Bt: bf16 [Ncols,K] row-major (transposed weights).
// OUT_MODE 0: bf16 out, row-major with leading dim ldo
// OUT_MODE 1: f32 out written NCHW-transposed: out[b*COUT*nhw + col*nhw + n], row=b*nhw+n
template <int OUT_MODE>
__global__ void gemm_wmma_kernel(const __bf16* __restrict__ A, const __bf16* __restrict__ Bt,
                                 const float* __restrict__ bias, int bias_n,
                                 const float* __restrict__ rowscale,
                                 void* __restrict__ outp, int K, int Ncols,
                                 int ldo, int relu, int nhw) {
  int lane = threadIdx.x & 31;
  int half = lane >> 4, lidx = lane & 15;
  int rowtile = blockIdx.x * 16;
  int ct0 = blockIdx.y * 32;
  int arow = rowtile + lidx;

  v8f acc0 = {}, acc1 = {};
  for (int k0 = 0; k0 < K; k0 += 32) {
    v16bf a  = load_frag_A(A, K, arow, k0, half);
    v16bf b0 = load_frag_Bt(Bt, K, ct0 + lidx, k0, half);
    v16bf b1 = load_frag_Bt(Bt, K, ct0 + 16 + lidx, k0, half);
    if (k0 + 32 < K) {
      __builtin_prefetch((const void*)(A + (size_t)arow * K + k0 + 32), 0, 1);
      __builtin_prefetch((const void*)(Bt + (size_t)(ct0 + lidx) * K + k0 + 32), 0, 1);
    }
    acc0 = WMMA_BF16(a, b0, acc0);
    acc1 = WMMA_BF16(a, b1, acc1);
  }

#pragma unroll
  for (int sub = 0; sub < 2; ++sub) {
    v8f acc = sub ? acc1 : acc0;
    int ocol = ct0 + sub * 16 + lidx;
#pragma unroll
    for (int r = 0; r < 8; ++r) {
      int orow = rowtile + r + 8 * half;
      float rs = rowscale ? rowscale[orow] : 1.0f;
      float v = acc[r];
      if (bias && ocol < bias_n) v += bias[ocol] * rs;
      if (relu) v = v > 0.f ? v : 0.f;
      if (OUT_MODE == 0) {
        ((__bf16*)outp)[(size_t)orow * ldo + ocol] = (__bf16)v;
      } else {
        int bb = orow / nhw, n = orow % nhw;
        ((float*)outp)[((size_t)bb * COUTc + ocol) * nhw + n] = v;
      }
    }
  }
}

// ---------------- K-predictor tail: logits + argmax ----------------
__global__ void kpred_argmax_kernel(const __bf16* __restrict__ mu,
                                    const float* __restrict__ kdec_w,
                                    const float* __restrict__ kdec_b,
                                    int* __restrict__ kint, float* __restrict__ kf) {
  int i = blockIdx.x * 256 + threadIdx.x;
  if (i >= MNODES) return;
  float m[32];
#pragma unroll
  for (int k = 0; k < 32; ++k) m[k] = (float)mu[(size_t)i * 32 + k];
  float bestv = -3.0e38f;
  int bestj = 0;
  for (int j = 0; j < Kk; ++j) {
    float l = kdec_b[j];
#pragma unroll
    for (int k = 0; k < 32; ++k) l += m[k] * kdec_w[k * Kk + j];
    if (l > bestv) { bestv = l; bestj = j; }
  }
  kint[i] = bestj;
  kf[i] = (float)bestj;
}

// ---------------- edge aggregation: concat(k*xi, sum_j xj - k*xi) ----------------
__global__ void edge_agg_kernel(const __bf16* __restrict__ feat, const int* __restrict__ nn,
                                const int* __restrict__ kint, __bf16* __restrict__ out) {
  int node = blockIdx.x;
  int c = threadIdx.x;  // 96 threads
  int kk = kint[node];
  float xi = (float)feat[(size_t)node * Cc + c];
  float s = 0.f;
  for (int e = 0; e < kk; ++e) {
    int j = nn[node * Kk + e];
    s += (float)feat[(size_t)j * Cc + c];
  }
  out[(size_t)node * (2 * Cc) + c]      = (__bf16)((float)kk * xi);
  out[(size_t)node * (2 * Cc) + Cc + c] = (__bf16)(s - (float)kk * xi);
}

// ---------------- launcher ----------------
extern "C" void kernel_launch(void* const* d_in, const int* in_sizes, int n_in,
                              void* d_out, int out_size, void* d_ws, size_t ws_size,
                              hipStream_t stream) {
  (void)in_sizes; (void)n_in; (void)out_size; (void)ws_size;
  const float* x      = (const float*)d_in[0];
  const float* kmap_w = (const float*)d_in[1];
  const float* kmap_b = (const float*)d_in[2];
  const float* kfc_w  = (const float*)d_in[3];
  const float* kfc_b  = (const float*)d_in[4];
  const float* kmu_w  = (const float*)d_in[5];
  const float* kmu_b  = (const float*)d_in[6];
  const float* kdec_w = (const float*)d_in[7];
  const float* kdec_b = (const float*)d_in[8];
  const float* ec1_w  = (const float*)d_in[9];
  const float* ec1_b  = (const float*)d_in[10];
  const float* ec2_w  = (const float*)d_in[11];
  const float* ec2_b  = (const float*)d_in[12];
  const float* fc_w   = (const float*)d_in[13];
  const float* fc_b   = (const float*)d_in[14];
  const float* io_w   = (const float*)d_in[15];
  const float* io_b   = (const float*)d_in[16];
  const float* up_w   = (const float*)d_in[17];
  const float* up_b   = (const float*)d_in[18];

  char* ws = (char*)d_ws;
  __bf16* XF   = (__bf16*)(ws + O_XF);
  float*  SQ   = (float*)(ws + O_SQ);
  int*    NN   = (int*)(ws + O_NN);
  int*    KI   = (int*)(ws + O_KI);
  float*  KF   = (float*)(ws + O_KF);
  __bf16* WKMAP= (__bf16*)(ws + O_WKMAP);
  __bf16* WKFC = (__bf16*)(ws + O_WKFC);
  __bf16* WKMU = (__bf16*)(ws + O_WKMU);
  __bf16* WEC1 = (__bf16*)(ws + O_WEC1);
  __bf16* WEC2 = (__bf16*)(ws + O_WEC2);
  __bf16* WFC  = (__bf16*)(ws + O_WFC);
  __bf16* WIO  = (__bf16*)(ws + O_WIO);
  __bf16* WUP  = (__bf16*)(ws + O_WUP);
  __bf16* HK   = (__bf16*)(ws + O_HK);
  __bf16* Hb   = (__bf16*)(ws + O_H);
  __bf16* MU   = (__bf16*)(ws + O_MU);
  __bf16* E1   = (__bf16*)(ws + O_E1);
  __bf16* H1   = (__bf16*)(ws + O_H1);
  __bf16* E2   = (__bf16*)(ws + O_E2);
  __bf16* H2   = (__bf16*)(ws + O_H2);
  __bf16* XA   = (__bf16*)(ws + O_XA);

  // 1) pack + norms
  {
    size_t tot = (size_t)Bn * Cc * Nn;
    pack_xf_kernel<<<(unsigned)((tot + 255) / 256), 256, 0, stream>>>(x, XF);
    sqnorm_kernel<<<MNODES, 32, 0, stream>>>(XF, SQ);
  }

  // 2) weight conversion: f32 [R,C] -> bf16 transposed+padded [Cp][Rp]
  auto padw = [&](const float* src, __bf16* dst, int R, int C_, int Rp, int Cp) {
    int tot = Rp * Cp;
    padT_bf16_kernel<<<(tot + 255) / 256, 256, 0, stream>>>(src, dst, R, C_, Rp, Cp);
  };
  padw(kmap_w, WKMAP, 96, 500, 96, 512);   // -> [512][96]
  padw(kfc_w,  WKFC, 500, 64, 512, 64);    // -> [64][512]
  padw(kmu_w,  WKMU,  64, 32,  64, 32);    // -> [32][64]
  padw(ec1_w,  WEC1, 192, 96, 192, 96);    // -> [96][192]
  padw(ec2_w,  WEC2, 192, 96, 192, 96);    // -> [96][192]
  padw(fc_w,   WFC,   96, 192, 96, 192);   // -> [192][96]
  padw(io_w,   WIO,   96, 192, 96, 192);   // -> [192][96]
  padw(up_w,   WUP,  384, 192, 384, 192);  // -> [192][384]

  // 3) kNN top-9 (Gram via WMMA)
  knn_topk_kernel<<<dim3(Nn / 16, Bn), 32, 0, stream>>>(XF, SQ, NN);

  dim3 blk(32);
  const int MT = MNODES / 16;  // 1568

  // 4) K-predictor
  gemm_wmma_kernel<0><<<dim3(MT, 512 / 32), blk, 0, stream>>>(
      XF, WKMAP, kmap_b, 500, nullptr, HK, 96, 512, 512, 0, 0);
  gemm_wmma_kernel<0><<<dim3(MT, 64 / 32), blk, 0, stream>>>(
      HK, WKFC, kfc_b, 64, nullptr, Hb, 512, 64, 64, 1, 0);
  gemm_wmma_kernel<0><<<dim3(MT, 32 / 32), blk, 0, stream>>>(
      Hb, WKMU, kmu_b, 32, nullptr, MU, 64, 32, 32, 0, 0);
  kpred_argmax_kernel<<<(MNODES + 255) / 256, 256, 0, stream>>>(MU, kdec_w, kdec_b, KI, KF);

  // 5) edge conv 1:  h1 = relu( concat(k*xi, sum xj - k*xi) @ ec1_w + k*ec1_b )
  edge_agg_kernel<<<MNODES, Cc, 0, stream>>>(XF, NN, KI, E1);
  gemm_wmma_kernel<0><<<dim3(MT, 96 / 32), blk, 0, stream>>>(
      E1, WEC1, ec1_b, 96, KF, H1, 192, 96, 96, 1, 0);

  // 6) edge conv 2 (no relu)
  edge_agg_kernel<<<MNODES, Cc, 0, stream>>>(H1, NN, KI, E2);
  gemm_wmma_kernel<0><<<dim3(MT, 96 / 32), blk, 0, stream>>>(
      E2, WEC2, ec2_b, 96, KF, H2, 192, 96, 96, 0, 0);

  // 7) xio -> XA[:, 0:192],  agg = h2@fc -> XA[:, 192:384]
  gemm_wmma_kernel<0><<<dim3(MT, 192 / 32), blk, 0, stream>>>(
      XF, WIO, io_b, 192, nullptr, XA, 96, 192, 384, 0, 0);
  gemm_wmma_kernel<0><<<dim3(MT, 192 / 32), blk, 0, stream>>>(
      H2, WFC, fc_b, 192, nullptr, XA + 192, 96, 192, 384, 0, 0);

  // 8) final: relu(XA @ up_w + up_b) written NCHW-transposed to d_out (f32)
  gemm_wmma_kernel<1><<<dim3(MT, 192 / 32), blk, 0, stream>>>(
      XA, WUP, up_b, 192, nullptr, d_out, 384, 192, 0, 1, Nn);
}
